// Gtransformerblock_89386859364866
// MI455X (gfx1250) — matrix operand order repaired and checked by reading, not compiled, not run on hardware
//
#include <hip/hip_runtime.h>

// HAN block on MI455X (gfx1250, wave32). All large GEMMs via
// v_wmma_f32_16x16x32_f16 (f16 A/B, f32 accum).
//
// Data-movement strategy (MI455X roofline):
//  * adj (134 MB f32, consumed by BOTH the graph-conv GEMM and the attention
//    mask) is packed to f16 once (exact: entries are 0/1). The 67 MB f16 copy
//    is resident in the 192 MB L2; HBM traffic ~200 MB vs ~540 MB.
//  * k_adjmm consumes the f16 adjacency through the Tensor Data Mover:
//    TENSOR_LOAD_TO_LDS DMAs 64x128 f16 tiles into LDS (double-buffered,
//    TENSORcnt-tracked, no VGPR transit), waves eat them as ds_load_b128
//    WMMA fragments while the next tile is in flight.
//  * Every other hot-loop WMMA operand is pre-packed f16 in fragment-friendly
//    layout (B-operands transposed) -> 16B loads, zero v_cvt in hot loops.
//  * Each wave computes all HID=64 output columns (4 accumulators).
//
// Exact simplifications (mathematically identical to the reference):
//  * gcn_norm: d = rowsum(softmax(..)) == 1 -> dinv factors are identity.
//  * semantic attention: softmax over a singleton axis -> beta == 1 -> p2=z0+z1.

#define N_NODES 4096
#define IN_DIM  512
#define HID     64
#define OUT_DIM 8
#define NHEAD   2

typedef _Float16 f16_t;
typedef __attribute__((ext_vector_type(16))) _Float16 v16h;
typedef __attribute__((ext_vector_type(8)))  _Float16 v8h;   // 16 B
typedef __attribute__((ext_vector_type(8)))  float    v8f;
typedef __attribute__((ext_vector_type(4)))  float    v4f;
typedef __attribute__((ext_vector_type(4)))  unsigned int v4u;
typedef __attribute__((ext_vector_type(8)))  int      v8i;
typedef __attribute__((ext_vector_type(4)))  int      v4i;

#if defined(__gfx1250__) && __has_builtin(__builtin_amdgcn_tensor_load_to_lds)
#define USE_TDM 1
#else
#define USE_TDM 0
#endif

union F16Frag { v16h v; v8h h[2]; };

__device__ __forceinline__ v8f wmma_f16(v16h a, v16h b, v8f c) {
  // (neg_a, A, neg_b, B, c_mod, C, reuse_a, reuse_b)
  return __builtin_amdgcn_wmma_f32_16x16x32_f16(false, a, false, b, (short)0, c,
                                                false, false);
}

// ---- Fragment loaders, contiguous k = 16*(lane/16)+s in A and B (wave32) ----

// A fragment 16x32 from f16 row-major (lda in elements): 2 x 16B per lane.
// lanes 0-15 : M=lane,    V0-7 = K 0..15
// lanes 16-31: M=lane-16, V0-7 = K 16..31
__device__ __forceinline__ v16h load_frag_a16(const f16_t* __restrict__ A, int lda) {
  const int lane = threadIdx.x & 31;
  const f16_t* row = A + (size_t)(lane & 15) * lda + (lane >> 4) * 16;
  F16Frag f;
  f.h[0] = *(const v8h*)(row);
  f.h[1] = *(const v8h*)(row + 8);
  return f.v;
}

// B fragment 32x16 where the operand is stored TRANSPOSED (n-major rows, ld):
// element (k,n) = Bt[n*ld + k]; each lane reads 16 contiguous f16 (2 x 16B).
// VGPR v: lanes 0-15 K=2v,2v+1 (N=lane); lanes 16-31 K=16+2v,17+2v.
__device__ __forceinline__ v16h load_frag_bt16(const f16_t* __restrict__ Bt, int ld) {
  const int lane = threadIdx.x & 31;
  const f16_t* row = Bt + (size_t)(lane & 15) * ld + (lane >> 4) * 16;
  F16Frag f;
  f.h[0] = *(const v8h*)(row);
  f.h[1] = *(const v8h*)(row + 8);
  return f.v;
}

// A fragment 16x32 from f32 row-major: 4 x 16B vector loads + packed converts.
__device__ __forceinline__ v16h load_frag_a_f32(const float* __restrict__ A, int lda) {
  const int lane = threadIdx.x & 31;
  const float* row = A + (size_t)(lane & 15) * lda + (lane >> 4) * 16;
  v4f x0 = *(const v4f*)(row);
  v4f x1 = *(const v4f*)(row + 4);
  v4f x2 = *(const v4f*)(row + 8);
  v4f x3 = *(const v4f*)(row + 12);
  v16h a;
#pragma unroll
  for (int j = 0; j < 4; ++j) {
    a[j]      = (_Float16)x0[j];
    a[4 + j]  = (_Float16)x1[j];
    a[8 + j]  = (_Float16)x2[j];
    a[12 + j] = (_Float16)x3[j];
  }
  return a;
}

// B fragment 32x16 from f32 row-major KxN (small weight matrices only).
__device__ __forceinline__ v16h load_frag_b_f32(const float* __restrict__ B, int ldb) {
  const int lane = threadIdx.x & 31;
  const int n = lane & 15;
  const int kb = (lane >> 4) * 16;
  v16h b;
#pragma unroll
  for (int v = 0; v < 8; ++v) {
    b[2 * v]     = (_Float16)B[(size_t)(kb + 2 * v) * ldb + n];
    b[2 * v + 1] = (_Float16)B[(size_t)(kb + 2 * v + 1) * ldb + n];
  }
  return b;
}

#if USE_TDM
// Issue a TDM load of a 64-row x 128-col f16 tile (row stride 4096 elements)
// from a 4096x4096 f16 tensor into LDS at byte offset lds_addr.
// D# per CDNA5 ISA ch.8: group0 {count,lds,global,type}, group1 {dims/strides}.
__device__ __forceinline__ void tdm_load_tile(const f16_t* gptr, unsigned lds_addr) {
  unsigned long long ga = (unsigned long long)(size_t)gptr;
  v4u g0;
  g0[0] = 1u;                                   // count=1 (valid descriptor)
  g0[1] = lds_addr;                             // lds_addr (bytes)
  g0[2] = (unsigned)ga;                         // global_addr[31:0]
  g0[3] = (unsigned)(ga >> 32) | (2u << 30);    // global_addr[56:32] | type=2
  v8i g1;
  g1[0] = (int)(1u << 16);        // workgroup_mask=0, data_size=1 (2 bytes)
  g1[1] = (int)(4096u << 16);     // tensor_dim0[15:0]=4096 in bits[79:64... 63:48]
  g1[2] = (int)(4096u << 16);     // tensor_dim0 hi=0, tensor_dim1[15:0]=4096
  g1[3] = (int)(128u << 16);      // tensor_dim1 hi=0, tile_dim0=128
  g1[4] = 64;                     // tile_dim1=64, tile_dim2=0
  g1[5] = 4096;                   // tensor_dim0_stride[31:0] (elements)
  g1[6] = 0;                      // stride hi, tensor_dim1_stride lo (unused, 2D)
  g1[7] = 0;
  v4i g2 = {0, 0, 0, 0};          // 2-D tensor: groups 2/3 unused
  v4i g3 = {0, 0, 0, 0};
#if defined(__clang_major__) && (__clang_major__ >= 23)
  v8i g4 = {0, 0, 0, 0, 0, 0, 0, 0};
  __builtin_amdgcn_tensor_load_to_lds(g0, g1, g2, g3, g4, 0);
#else
  __builtin_amdgcn_tensor_load_to_lds(g0, g1, g2, g3, 0);
#endif
}
#endif

// ---- Kernel 0: pack adj f32 -> f16 (exact: entries are 0/1) ----
__global__ __launch_bounds__(256) void k_cvt(const float* __restrict__ src,
                                             f16_t* __restrict__ dst) {
  size_t i = ((size_t)blockIdx.x * 256 + threadIdx.x) * 8;
  v4f x0 = *(const v4f*)(src + i);
  v4f x1 = *(const v4f*)(src + i + 4);
  v8h o;
#pragma unroll
  for (int j = 0; j < 4; ++j) { o[j] = (_Float16)x0[j]; o[4 + j] = (_Float16)x1[j]; }
  *(v8h*)(dst + i) = o;
}

// ---- Kernel 1: hWt[head] = (h @ W1[head])^T, stored f16 [HID][N] ----
__global__ __launch_bounds__(128) void k_hw(const float* __restrict__ h,
                                            const float* __restrict__ W1,
                                            f16_t* __restrict__ hWt) {
  const int head = blockIdx.z;
  const int m0 = (blockIdx.x * 4 + threadIdx.y) * 16;
  const float* Wh = W1 + (size_t)head * IN_DIM * HID;
  v8f acc[4] = {};
  for (int k0 = 0; k0 < IN_DIM; k0 += 32) {
    v16h a = load_frag_a_f32(h + (size_t)m0 * IN_DIM + k0, IN_DIM);
#pragma unroll
    for (int t = 0; t < 4; ++t)
      acc[t] = wmma_f16(a, load_frag_b_f32(Wh + (size_t)k0 * HID + t * 16, HID), acc[t]);
  }
  const int lane = threadIdx.x & 31;
  const int n = lane & 15;
  const int mb = (lane >> 4) * 8;
  f16_t* D = hWt + (size_t)head * HID * N_NODES;
#pragma unroll
  for (int t = 0; t < 4; ++t)
#pragma unroll
    for (int r = 0; r < 8; ++r)
      D[(size_t)(t * 16 + n) * N_NODES + m0 + mb + r] = (_Float16)acc[t][r];
}

// ---- Kernel 2: x16[head] = relu(adj16 @ hW + b1), f16 [N][HID] ----
// A-side (adj16) staged through LDS by the Tensor Data Mover, double-buffered.
__global__ __launch_bounds__(128) void k_adjmm(const f16_t* __restrict__ adj16,
                                               const f16_t* __restrict__ hWt,
                                               const float* __restrict__ b1,
                                               f16_t* __restrict__ x16) {
  const int head = blockIdx.z;
  const int wid = threadIdx.y;
  const int m0 = blockIdx.x * 64 + wid * 16;
  const f16_t* A = adj16 + (size_t)head * N_NODES * N_NODES;
  const f16_t* Bt = hWt + (size_t)head * HID * N_NODES;
  v8f acc[4] = {};
#if USE_TDM
  __shared__ f16_t sT[2][64][128];   // 2 x 16 KB tile buffers
  const f16_t* Ablk = A + (size_t)(blockIdx.x * 64) * N_NODES;
  const unsigned lds0 = (unsigned)(size_t)&sT[0][0][0];
  if (wid == 0) tdm_load_tile(Ablk, lds0);             // T0 in flight
  for (int i = 0; i < N_NODES / 128; ++i) {
    if (wid == 0) {
      if (i + 1 < N_NODES / 128) {
        tdm_load_tile(Ablk + (size_t)(i + 1) * 128, lds0 + ((i + 1) & 1) * 16384);
        __builtin_amdgcn_s_wait_tensorcnt(1);          // T_i done, T_{i+1} flying
      } else {
        __builtin_amdgcn_s_wait_tensorcnt(0);
      }
    }
    __syncthreads();                                   // tile i visible to all waves
    const f16_t* tile = &sT[i & 1][wid * 16][0];
    const int k0 = i * 128;
#pragma unroll
    for (int kk = 0; kk < 128; kk += 32) {
      v16h a = load_frag_a16(tile + kk, 128);          // ds_load_b128 fragments
#pragma unroll
      for (int t = 0; t < 4; ++t)
        acc[t] = wmma_f16(
            a, load_frag_bt16(Bt + (size_t)(t * 16) * N_NODES + k0 + kk, N_NODES),
            acc[t]);
    }
    __syncthreads();                                   // buffer reusable for T_{i+2}
  }
#else
  for (int k0 = 0; k0 < N_NODES; k0 += 32) {
    __builtin_prefetch(A + (size_t)m0 * N_NODES + k0 + 512, 0, 1);
    v16h a = load_frag_a16(A + (size_t)m0 * N_NODES + k0, N_NODES);
#pragma unroll
    for (int t = 0; t < 4; ++t)
      acc[t] = wmma_f16(a, load_frag_bt16(Bt + (size_t)(t * 16) * N_NODES + k0, N_NODES),
                        acc[t]);
  }
#endif
  const int lane = threadIdx.x & 31;
  const int n = lane & 15;
  const int mb = (lane >> 4) * 8;
  f16_t* D = x16 + (size_t)head * N_NODES * HID;
#pragma unroll
  for (int t = 0; t < 4; ++t) {
    const float bias = b1[head * HID + t * 16 + n];
#pragma unroll
    for (int r = 0; r < 8; ++r) {
      float v = acc[t][r] + bias;
      D[(size_t)(m0 + mb + r) * HID + t * 16 + n] = (_Float16)(v > 0.f ? v : 0.f);
    }
  }
}

// ---- Kernel 3: Q16/K16 row-major [N][HID], Vt16 transposed [HID][N] ----
__global__ __launch_bounds__(128) void k_qkv(const f16_t* __restrict__ x16,
                                             const float* __restrict__ Wq, const float* __restrict__ bq,
                                             const float* __restrict__ Wk, const float* __restrict__ bk,
                                             const float* __restrict__ Wv, const float* __restrict__ bv,
                                             f16_t* __restrict__ Q16, f16_t* __restrict__ K16,
                                             f16_t* __restrict__ Vt16) {
  const int which = blockIdx.z % 3;
  const int head = blockIdx.z / 3;
  const float* W = (which == 0) ? Wq : (which == 1) ? Wk : Wv;
  const float* bb = (which == 0) ? bq : (which == 1) ? bk : bv;
  const int m0 = (blockIdx.x * 4 + threadIdx.y) * 16;
  const f16_t* X = x16 + (size_t)head * N_NODES * HID;
  const float* Wh = W + (size_t)head * HID * HID;
  v8f acc[4] = {};
#pragma unroll
  for (int k0 = 0; k0 < HID; k0 += 32) {
    v16h a = load_frag_a16(X + (size_t)m0 * HID + k0, HID);
#pragma unroll
    for (int t = 0; t < 4; ++t)
      acc[t] = wmma_f16(a, load_frag_b_f32(Wh + (size_t)k0 * HID + t * 16, HID), acc[t]);
  }
  const int lane = threadIdx.x & 31;
  const int n = lane & 15;
  const int mb = (lane >> 4) * 8;
  if (which == 2) {  // V stored transposed [HID][N] for contiguous B-fragments
    f16_t* D = Vt16 + (size_t)head * HID * N_NODES;
#pragma unroll
    for (int t = 0; t < 4; ++t) {
      const float bias = bb[head * HID + t * 16 + n];
#pragma unroll
      for (int r = 0; r < 8; ++r)
        D[(size_t)(t * 16 + n) * N_NODES + m0 + mb + r] = (_Float16)(acc[t][r] + bias);
    }
  } else {
    f16_t* D = ((which == 0) ? Q16 : K16) + (size_t)head * N_NODES * HID;
#pragma unroll
    for (int t = 0; t < 4; ++t) {
      const float bias = bb[head * HID + t * 16 + n];
#pragma unroll
      for (int r = 0; r < 8; ++r)
        D[(size_t)(m0 + mb + r) * HID + t * 16 + n] = (_Float16)(acc[t][r] + bias);
    }
  }
}

// ---- Kernel 4: fused masked attention (flash-style, dense softmax rows) ----
// z = relu(softmax(adj * (Q K^T)) @ V); gcn_norm is the identity (see header).
__global__ __launch_bounds__(128) void k_attn(const f16_t* __restrict__ adj16,
                                              const f16_t* __restrict__ Q16,
                                              const f16_t* __restrict__ K16,
                                              const f16_t* __restrict__ Vt16,
                                              float* __restrict__ Z) {
  __shared__ f16_t sP[4][16 * 32];
  const int head = blockIdx.z;
  const int wid = threadIdx.y;
  const int m0 = (blockIdx.x * 4 + wid) * 16;
  const int lane = threadIdx.x & 31;
  const int n = lane & 15;
  const int mb = (lane >> 4) * 8;
  const f16_t* Ah = adj16 + (size_t)head * N_NODES * N_NODES;
  const f16_t* Qh = Q16 + (size_t)head * N_NODES * HID;
  const f16_t* Kh = K16 + (size_t)head * N_NODES * HID;
  const f16_t* Vt = Vt16 + (size_t)head * HID * N_NODES;

  // Q row-tile fragments (HID=64 -> two K-chunks), loaded once per wave.
  v16h qa0 = load_frag_a16(Qh + (size_t)m0 * HID + 0, HID);
  v16h qa1 = load_frag_a16(Qh + (size_t)m0 * HID + 32, HID);

  v8f o[4] = {};
  float mrun[8], lrun[8];
#pragma unroll
  for (int r = 0; r < 8; ++r) { mrun[r] = -3.0e38f; lrun[r] = 0.f; }

  for (int c0 = 0; c0 < N_NODES; c0 += 32) {
    __builtin_prefetch(Ah + (size_t)(m0 + mb) * N_NODES + c0 + 512, 0, 1);
    // S = Q @ K^T for this 16x32 block; K16 is row-major [N][HID] so the
    // transposed operand gives contiguous per-lane loads.
    v8f s0 = {}, s1 = {};
    s0 = wmma_f16(qa0, load_frag_bt16(Kh + (size_t)(c0 + 0) * HID + 0, HID), s0);
    s0 = wmma_f16(qa1, load_frag_bt16(Kh + (size_t)(c0 + 0) * HID + 32, HID), s0);
    s1 = wmma_f16(qa0, load_frag_bt16(Kh + (size_t)(c0 + 16) * HID + 0, HID), s1);
    s1 = wmma_f16(qa1, load_frag_bt16(Kh + (size_t)(c0 + 16) * HID + 32, HID), s1);
    // A_tilde = adj * (Q K^T); adj entries are exactly 0/1.
#pragma unroll
    for (int r = 0; r < 8; ++r) {
      s0[r] *= (float)Ah[(size_t)(m0 + mb + r) * N_NODES + c0 + n];
      s1[r] *= (float)Ah[(size_t)(m0 + mb + r) * N_NODES + c0 + 16 + n];
    }
    // Online softmax over the full dense row (zeros included for non-edges).
    // Row m = mb + r lives on one 16-lane half-wave -> shfl-xor reductions.
#pragma unroll
    for (int r = 0; r < 8; ++r) {
      float mx = fmaxf(s0[r], s1[r]);
#pragma unroll
      for (int d = 1; d < 16; d <<= 1) mx = fmaxf(mx, __shfl_xor(mx, d, 16));
      float mnew = fmaxf(mrun[r], mx);
      float corr = __expf(mrun[r] - mnew);
      float p0 = __expf(s0[r] - mnew);
      float p1 = __expf(s1[r] - mnew);
      float rs = p0 + p1;
#pragma unroll
      for (int d = 1; d < 16; d <<= 1) rs += __shfl_xor(rs, d, 16);
      lrun[r] = lrun[r] * corr + rs;
      mrun[r] = mnew;
      o[0][r] *= corr; o[1][r] *= corr; o[2][r] *= corr; o[3][r] *= corr;
      // Re-layout P (C-layout) -> A-layout via LDS (f16, b128 reload).
      sP[wid][(mb + r) * 32 + n] = (_Float16)p0;
      sP[wid][(mb + r) * 32 + 16 + n] = (_Float16)p1;
    }
    __syncthreads();
    v16h pf = load_frag_a16(&sP[wid][0], 32);
    __syncthreads();
#pragma unroll
    for (int t = 0; t < 4; ++t)
      o[t] = wmma_f16(pf, load_frag_bt16(Vt + (size_t)(t * 16) * N_NODES + c0, N_NODES),
                      o[t]);
  }
  // rowsum(att)==1 -> dinv factors identity; just 1/l, relu, store f32.
  float* Zh = Z + (size_t)head * N_NODES * HID + (size_t)m0 * HID;
#pragma unroll
  for (int r = 0; r < 8; ++r) {
    float inv = 1.f / lrun[r];
#pragma unroll
    for (int t = 0; t < 4; ++t) {
      float v = o[t][r] * inv;
      Zh[(size_t)(mb + r) * HID + t * 16 + n] = v > 0.f ? v : 0.f;
    }
  }
}

// ---- Kernel 5: p2 = z0 + z1 (beta == 1), logits, row softmax ----
__global__ __launch_bounds__(256) void k_pred(const float* __restrict__ Z,
                                              const float* __restrict__ predW,
                                              const float* __restrict__ predb,
                                              float* __restrict__ out) {
  int i = blockIdx.x * blockDim.x + threadIdx.x;
  if (i >= N_NODES) return;
  float logits[OUT_DIM];
#pragma unroll
  for (int o = 0; o < OUT_DIM; ++o) logits[o] = predb[o];
  for (int c = 0; c < HID; ++c) {
    float p = Z[(size_t)i * HID + c] + Z[(size_t)N_NODES * HID + (size_t)i * HID + c];
#pragma unroll
    for (int o = 0; o < OUT_DIM; ++o) logits[o] += p * predW[c * OUT_DIM + o];
  }
  float mx = logits[0];
#pragma unroll
  for (int o = 1; o < OUT_DIM; ++o) mx = fmaxf(mx, logits[o]);
  float s = 0.f;
#pragma unroll
  for (int o = 0; o < OUT_DIM; ++o) { logits[o] = __expf(logits[o] - mx); s += logits[o]; }
  float inv = 1.f / s;
#pragma unroll
  for (int o = 0; o < OUT_DIM; ++o) out[(size_t)i * OUT_DIM + o] = logits[o] * inv;
}

extern "C" void kernel_launch(void* const* d_in, const int* in_sizes, int n_in,
                              void* d_out, int out_size, void* d_ws, size_t ws_size,
                              hipStream_t stream) {
  const float* h   = (const float*)d_in[0];
  const float* adj = (const float*)d_in[1];
  const float* W1  = (const float*)d_in[2];
  const float* b1  = (const float*)d_in[3];
  const float* Wq  = (const float*)d_in[4];
  const float* bq  = (const float*)d_in[5];
  const float* Wk  = (const float*)d_in[6];
  const float* bk  = (const float*)d_in[7];
  const float* Wv  = (const float*)d_in[8];
  const float* bv  = (const float*)d_in[9];
  // d_in[10..12] = semantic attention params: provably no effect on the output
  // (softmax over a singleton axis -> beta == 1 for every head).
  const float* predW = (const float*)d_in[13];
  const float* predb = (const float*)d_in[14];

  const size_t HNN = (size_t)NHEAD * N_NODES * N_NODES;  // 33.5M elems
  const size_t HNH = (size_t)NHEAD * N_NODES * HID;      // 524288 elems
  f16_t* adj16 = (f16_t*)d_ws;          // 64 MiB, L2-resident working set
  f16_t* hWt   = adj16 + HNN;           // [H][HID][N] 1 MiB
  f16_t* x16   = hWt + HNH;             // [H][N][HID] 1 MiB
  f16_t* Q16   = x16 + HNH;             // [H][N][HID] 1 MiB
  f16_t* K16   = Q16 + HNH;             // [H][N][HID] 1 MiB
  f16_t* Vt16  = K16 + HNH;             // [H][HID][N] 1 MiB
  float* Z     = (float*)(Vt16 + HNH);  // [H][N][HID] f32, 2 MiB

  dim3 blk(32, 4);
  k_cvt  <<<dim3((unsigned)(HNN / (8 * 256))), dim3(256), 0, stream>>>(adj, adj16);
  k_hw   <<<dim3(N_NODES / 64, 1, NHEAD),     blk, 0, stream>>>(h, W1, hWt);
  k_adjmm<<<dim3(N_NODES / 64, 1, NHEAD),     blk, 0, stream>>>(adj16, hWt, b1, x16);
  k_qkv  <<<dim3(N_NODES / 64, 1, NHEAD * 3), blk, 0, stream>>>(x16, Wq, bq, Wk, bk,
                                                                Wv, bv, Q16, K16, Vt16);
  k_attn <<<dim3(N_NODES / 64, 1, NHEAD),     blk, 0, stream>>>(adj16, Q16, K16, Vt16, Z);
  k_pred <<<dim3(N_NODES / 256), dim3(256),   0, stream>>>(Z, predW, predb,
                                                           (float*)d_out);
}
